// MultiHeadAttention_30889404793243
// MI455X (gfx1250) — matrix operand order repaired and checked by reading, not compiled, not run on hardware
//
#include <hip/hip_runtime.h>

#define D_MODEL 768
#define N_HEADS 12
#define HEAD_DIM 64
#define BATCH 4
#define SEQ 2048
#define ROWS (BATCH * SEQ)     // 8192
#define BH (BATCH * N_HEADS)   // 48
#define NK (D_MODEL / 32)      // 24 K-chunks of 32

typedef __attribute__((ext_vector_type(16))) __bf16 bf16x16;
typedef __attribute__((ext_vector_type(8)))  float  f32x8;
typedef unsigned short u16;

union FragU { bf16x16 v; uint4 q[2]; };

__device__ __forceinline__ u16 f2bf(float f) {
  unsigned int u = __float_as_uint(f);
  u += 0x7fffu + ((u >> 16) & 1u);           // round-to-nearest-even
  return (u16)(u >> 16);
}

__device__ __forceinline__ bf16x16 ld_frag(const u16* p0, const u16* p1) {
  FragU f;
  f.q[0] = *(const uint4*)p0;
  f.q[1] = *(const uint4*)p1;
  return f.v;
}

__device__ __forceinline__ f32x8 wmma_bf16(bf16x16 a, bf16x16 b, f32x8 c) {
  // D = A(16x32 bf16) * B(32x16 bf16) + C(16x16 f32)
  return __builtin_amdgcn_wmma_f32_16x16x32_bf16(
      false, a, false, b, (short)0, c, false, false);
}

// gfx1250 async copy: global -> LDS, 16B per lane, tracked by ASYNCcnt.
__device__ __forceinline__ void async_load_b128(const u16* g, u16* l) {
  unsigned lds = (unsigned)(uintptr_t)l;              // low 32 bits = LDS offset
  unsigned long long ga = (unsigned long long)(uintptr_t)g;
  asm volatile("global_load_async_to_lds_b128 %0, %1, off"
               :: "v"(lds), "v"(ga) : "memory");
}

// ---------------------------------------------------------------- conversion
__global__ void cvt_f32_bf16(const float* __restrict__ in, u16* __restrict__ out, int n) {
  int i = blockIdx.x * blockDim.x + threadIdx.x;
  if (i < n) out[i] = f2bf(in[i]);
}

// --------------------------------------------------- wave 32x64 GEMM (LDS-B)
// Y[m, n] = sum_d A[m, d] * W[n, d]. Block = 128 threads (4 waves) covering a
// 128x64 output tile; the 64x32 weight chunk is async-staged into a
// double-buffered LDS tile shared by all 4 waves.
__device__ __forceinline__ void gemm32x64_lds(const u16* __restrict__ A,
                                              const u16* __restrict__ W,
                                              u16* __restrict__ wbuf,  // [2][64*32]
                                              int mBase, int nBase,
                                              int tid, f32x8 c[2][4]) {
  const int lane = tid & 31;
  const int rsel = lane & 15, half = lane >> 4;
  #pragma unroll
  for (int mi = 0; mi < 2; ++mi)
    #pragma unroll
    for (int t = 0; t < 4; ++t) { f32x8 z = {}; c[mi][t] = z; }

  // cooperative async staging: thread -> (row = tid/2, 16-element run)
  const int row = tid >> 1, e0 = (tid & 1) * 16;
  const u16* gW = W + (size_t)(nBase + row) * D_MODEL + e0;
  u16* lW = wbuf + row * 32 + e0;

  // prefetch chunk 0 into buffer 0
  async_load_b128(gW, lW);
  async_load_b128(gW + 8, lW + 8);

  const u16* aRow0 = A + (size_t)(mBase + rsel) * D_MODEL + half * 8;
  const u16* aRow1 = aRow0 + (size_t)16 * D_MODEL;

  #pragma unroll 1
  for (int kc = 0; kc < NK; ++kc) {
    const int cur = kc & 1;
    // A fragments for this chunk: independent of LDS, overlap the async wait
    const int k = kc * 32;
    const bf16x16 a0 = ld_frag(aRow0 + k, aRow0 + k + 16);
    const bf16x16 a1 = ld_frag(aRow1 + k, aRow1 + k + 16);
    if (kc + 1 < NK) {
      // issue prefetch of chunk kc+1 before waiting on chunk kc
      const u16* g = gW + (kc + 1) * 32;
      u16* l = wbuf + (cur ^ 1) * (64 * 32) + row * 32 + e0;
      async_load_b128(g, l);
      async_load_b128(g + 8, l + 8);
      asm volatile("s_wait_asynccnt 0x2" ::: "memory");  // chunk kc landed
    } else {
      asm volatile("s_wait_asynccnt 0x0" ::: "memory");
    }
    __syncthreads();                                     // tile visible to WGP

    const u16* lb = wbuf + cur * (64 * 32);
    // preload all 4 B fragments so the 8 WMMAs are not serialized on DScnt
    bf16x16 bfr[4];
    #pragma unroll
    for (int t = 0; t < 4; ++t) {
      const u16* bp = lb + (t * 16 + rsel) * 32 + half * 8;
      bfr[t] = ld_frag(bp, bp + 16);
    }
    #pragma unroll
    for (int t = 0; t < 4; ++t) {
      c[0][t] = wmma_bf16(a0, bfr[t], c[0][t]);
      c[1][t] = wmma_bf16(a1, bfr[t], c[1][t]);
    }
    __syncthreads();                  // all reads done before buffer overwrite
  }
}

// ------------------------------------------------------------- QKV projection
// blockIdx.z: 0=Q (scaled by 1/sqrt(Dh)), 1=K, 2=V (stored transposed).
__global__ void __launch_bounds__(128, 1)
qkv_gemm(const u16* __restrict__ Xb,
         const u16* __restrict__ Wqb, const u16* __restrict__ Wkb,
         const u16* __restrict__ Wvb,
         const float* __restrict__ bq, const float* __restrict__ bk,
         const float* __restrict__ bv,
         u16* __restrict__ Qb, u16* __restrict__ Kb,
         u16* __restrict__ Vtb) {
  __shared__ u16 wbuf[2 * 64 * 32];   // 8KB double-buffered weight tile
  const int tid = threadIdx.x;
  const int lane = tid & 31;
  const int wave = tid >> 5;
  const int mBase = blockIdx.x * 128 + wave * 32;
  const int nBase = blockIdx.y * 64;
  const int mode = blockIdx.z;
  const u16* W = (mode == 0) ? Wqb : (mode == 1) ? Wkb : Wvb;
  const float* bias = (mode == 0) ? bq : (mode == 1) ? bk : bv;

  f32x8 c[2][4];
  gemm32x64_lds(Xb, W, wbuf, mBase, nBase, tid, c);

  const int rsel = lane & 15, half = lane >> 4;
  #pragma unroll
  for (int mi = 0; mi < 2; ++mi) {
    #pragma unroll
    for (int t = 0; t < 4; ++t) {
      const int nCol = nBase + t * 16 + rsel;
      const int h = nCol >> 6, dh = nCol & 63;
      const float bb = bias[nCol];
      #pragma unroll
      for (int r = 0; r < 8; ++r) {
        const int m = mBase + mi * 16 + half * 8 + r;   // global token row
        const int b = m >> 11, s = m & (SEQ - 1);
        float val = c[mi][t][r] + bb;
        if (mode == 0) val *= 0.125f;                   // 1/sqrt(HEAD_DIM)
        const u16 hv = f2bf(val);
        const size_t bh = (size_t)(b * N_HEADS + h);
        if (mode == 2)      Vtb[(bh * HEAD_DIM + dh) * SEQ + s] = hv;  // (Dh,S)
        else if (mode == 0) Qb[(bh * SEQ + s) * HEAD_DIM + dh] = hv;   // (S,Dh)
        else                Kb[(bh * SEQ + s) * HEAD_DIM + dh] = hv;
      }
    }
  }
}

// ---------------------------------------------------- flash attention (wave)
__global__ void attn_kernel(const u16* __restrict__ Qb, const u16* __restrict__ Kb,
                            const u16* __restrict__ Vtb, const int* __restrict__ mask,
                            u16* __restrict__ Ctxb) {
  __shared__ u16 ldsP[4 * 16 * 32];                 // 1KB staging per wave
  const int lane = threadIdx.x & 31;
  const int wave = threadIdx.x >> 5;
  const int g = blockIdx.x * 4 + wave;
  const int bh = g >> 7;                            // / (SEQ/16)
  const int qt = g & 127;
  const int qBase = qt * 16;
  const int b = bh / N_HEADS;
  const int rsel = lane & 15, half = lane >> 4;

  const u16* Qp = Qb + (size_t)bh * SEQ * HEAD_DIM;
  const u16* Kp = Kb + (size_t)bh * SEQ * HEAD_DIM;
  const u16* Vp = Vtb + (size_t)bh * HEAD_DIM * SEQ;
  const int* pm = mask + b * SEQ;
  u16* lds = ldsP + wave * 512;

  // Q tile 16x64 as two A fragments (d 0..31 / 32..63); scale pre-folded.
  const u16* qRowP = Qp + (size_t)(qBase + rsel) * HEAD_DIM + half * 8;
  const bf16x16 aQ0 = ld_frag(qRowP, qRowP + 16);
  const bf16x16 aQ1 = ld_frag(qRowP + 32, qRowP + 48);

  f32x8 o[4];
  #pragma unroll
  for (int t = 0; t < 4; ++t) { f32x8 z = {}; o[t] = z; }
  float mst[8], lst[8];
  #pragma unroll
  for (int r = 0; r < 8; ++r) { mst[r] = -3.0e38f; lst[r] = 0.f; }
  const int rowBase = qBase + half * 8;             // first of this lane's 8 rows
  const int kEnd = qBase + 16;                      // causal bound (exclusive)

  for (int kb = 0; kb < kEnd; kb += 32) {
    #pragma unroll
    for (int t = 0; t < 2; ++t) {
      const int kcol = kb + t * 16 + rsel;          // this lane's key column
      const u16* kp = Kp + (size_t)kcol * HEAD_DIM + half * 8;
      const bf16x16 b0 = ld_frag(kp, kp + 16);      // d 0..31
      const bf16x16 b1 = ld_frag(kp + 32, kp + 48); // d 32..63
      f32x8 s = {};
      s = wmma_bf16(aQ0, b0, s);
      s = wmma_bf16(aQ1, b1, s);

      const int pmv = pm[kcol];
      float sv[8];
      #pragma unroll
      for (int r = 0; r < 8; ++r) {
        const bool ok = (pmv != 0) && (kcol <= rowBase + r);
        sv[r] = ok ? s[r] : -1.0e30f;
      }
      // row-max across the 16 lanes of this half
      float tmax[8];
      #pragma unroll
      for (int r = 0; r < 8; ++r) tmax[r] = sv[r];
      #pragma unroll
      for (int off = 8; off >= 1; off >>= 1)
        #pragma unroll
        for (int r = 0; r < 8; ++r)
          tmax[r] = fmaxf(tmax[r], __shfl_xor(tmax[r], off, 32));

      float pv[8], rsum[8], fac[8];
      #pragma unroll
      for (int r = 0; r < 8; ++r) {
        const float mn = fmaxf(mst[r], tmax[r]);
        fac[r] = __expf(mst[r] - mn);
        pv[r] = __expf(sv[r] - mn);
        mst[r] = mn;
        rsum[r] = pv[r];
      }
      #pragma unroll
      for (int off = 8; off >= 1; off >>= 1)
        #pragma unroll
        for (int r = 0; r < 8; ++r)
          rsum[r] += __shfl_xor(rsum[r], off, 32);
      #pragma unroll
      for (int r = 0; r < 8; ++r) lst[r] = lst[r] * fac[r] + rsum[r];
      #pragma unroll
      for (int t2 = 0; t2 < 4; ++t2)
        #pragma unroll
        for (int r = 0; r < 8; ++r) o[t2][r] = o[t2][r] * fac[r];
      // stage P (C-layout) -> LDS [row][key 0..31] as bf16
      const int col = t * 16 + rsel;
      #pragma unroll
      for (int r = 0; r < 8; ++r) lds[(half * 8 + r) * 32 + col] = f2bf(pv[r]);
    }
    // reload P as A fragment (same-wave LDS ops are in-order)
    const u16* lp = lds + rsel * 32 + half * 8;
    bf16x16 aP;
    { FragU f; f.q[0] = *(const uint4*)lp; f.q[1] = *(const uint4*)(lp + 16); aP = f.v; }
    // O(16x64) += P(16x32) * V(32x64), V^T rows contiguous along keys.
    // Preload all 4 V fragments so the WMMAs issue back-to-back.
    bf16x16 bV[4];
    #pragma unroll
    for (int t2 = 0; t2 < 4; ++t2) {
      const u16* vp = Vp + (size_t)(t2 * 16 + rsel) * SEQ + kb + half * 8;
      bV[t2] = ld_frag(vp, vp + 16);
    }
    #pragma unroll
    for (int t2 = 0; t2 < 4; ++t2) o[t2] = wmma_bf16(aP, bV[t2], o[t2]);
  }

  // epilogue: O /= l, write ctx (B,S,H*Dh) bf16
  const int h = bh % N_HEADS;
  #pragma unroll
  for (int t2 = 0; t2 < 4; ++t2) {
    const int col = h * HEAD_DIM + t2 * 16 + rsel;
    #pragma unroll
    for (int r = 0; r < 8; ++r) {
      const int m = rowBase + r;
      const size_t row = (size_t)b * SEQ + m;
      Ctxb[row * D_MODEL + col] = f2bf(o[t2][r] / lst[r]);
    }
  }
}

// ---------------------------------------------------------- output projection
__global__ void __launch_bounds__(128, 1)
out_gemm(const u16* __restrict__ Ctxb, const u16* __restrict__ Wob,
         const float* __restrict__ bo, float* __restrict__ out) {
  __shared__ u16 wbuf[2 * 64 * 32];
  const int tid = threadIdx.x;
  const int lane = tid & 31;
  const int wave = tid >> 5;
  const int mBase = blockIdx.x * 128 + wave * 32;
  const int nBase = blockIdx.y * 64;
  f32x8 c[2][4];
  gemm32x64_lds(Ctxb, Wob, wbuf, mBase, nBase, tid, c);
  const int rsel = lane & 15, half = lane >> 4;
  #pragma unroll
  for (int mi = 0; mi < 2; ++mi) {
    #pragma unroll
    for (int t = 0; t < 4; ++t) {
      const int nCol = nBase + t * 16 + rsel;
      const float bb = bo[nCol];
      #pragma unroll
      for (int r = 0; r < 8; ++r) {
        const int m = mBase + mi * 16 + half * 8 + r;
        out[(size_t)m * D_MODEL + nCol] = c[mi][t][r] + bb;
      }
    }
  }
}

// -------------------------------------------------------------------- launch
extern "C" void kernel_launch(void* const* d_in, const int* in_sizes, int n_in,
                              void* d_out, int out_size, void* d_ws, size_t ws_size,
                              hipStream_t stream) {
  (void)in_sizes; (void)n_in; (void)out_size; (void)ws_size;
  const float* X    = (const float*)d_in[0];
  const int*   mask = (const int*)d_in[1];
  const float* Wq = (const float*)d_in[2];
  const float* bq = (const float*)d_in[3];
  const float* Wk = (const float*)d_in[4];
  const float* bk = (const float*)d_in[5];
  const float* Wv = (const float*)d_in[6];
  const float* bv = (const float*)d_in[7];
  const float* Wo = (const float*)d_in[8];
  const float* bo = (const float*)d_in[9];
  float* out = (float*)d_out;

  char* ws = (char*)d_ws;
  size_t off = 0;
  auto take = [&](size_t bytes) -> char* {
    char* p = ws + off;
    off += (bytes + 255) & ~(size_t)255;
    return p;
  };
  u16* Xb  = (u16*)take((size_t)ROWS * D_MODEL * 2);
  u16* Wqb = (u16*)take((size_t)D_MODEL * D_MODEL * 2);
  u16* Wkb = (u16*)take((size_t)D_MODEL * D_MODEL * 2);
  u16* Wvb = (u16*)take((size_t)D_MODEL * D_MODEL * 2);
  u16* Wob = (u16*)take((size_t)D_MODEL * D_MODEL * 2);
  u16* Qb  = (u16*)take((size_t)BH * SEQ * HEAD_DIM * 2);
  u16* Kb  = (u16*)take((size_t)BH * SEQ * HEAD_DIM * 2);
  u16* Vtb = (u16*)take((size_t)BH * HEAD_DIM * SEQ * 2);
  u16* Ctxb = (u16*)take((size_t)ROWS * D_MODEL * 2);

  const int nX = ROWS * D_MODEL;
  const int nW = D_MODEL * D_MODEL;
  cvt_f32_bf16<<<(nX + 255) / 256, 256, 0, stream>>>(X, Xb, nX);
  cvt_f32_bf16<<<(nW + 255) / 256, 256, 0, stream>>>(Wq, Wqb, nW);
  cvt_f32_bf16<<<(nW + 255) / 256, 256, 0, stream>>>(Wk, Wkb, nW);
  cvt_f32_bf16<<<(nW + 255) / 256, 256, 0, stream>>>(Wv, Wvb, nW);
  cvt_f32_bf16<<<(nW + 255) / 256, 256, 0, stream>>>(Wo, Wob, nW);

  dim3 gQKV(ROWS / 128, D_MODEL / 64, 3);
  qkv_gemm<<<gQKV, 128, 0, stream>>>(Xb, Wqb, Wkb, Wvb, bq, bk, bv, Qb, Kb, Vtb);

  attn_kernel<<<(BH * (SEQ / 16)) / 4, 128, 0, stream>>>(Qb, Kb, Vtb, mask, Ctxb);

  out_gemm<<<dim3(ROWS / 128, D_MODEL / 64), 128, 0, stream>>>(Ctxb, Wob, bo, out);
}
